// MultiHeadAttention_88218628259892
// MI455X (gfx1250) — matrix operand order repaired and checked by reading, not compiled, not run on hardware
//
#include <hip/hip_runtime.h>
#include <cstdint>

typedef __bf16 bf16;
typedef __attribute__((ext_vector_type(16))) __bf16 v16bf;
typedef __attribute__((ext_vector_type(8)))  float   v8f;
typedef int v4i_vs __attribute__((vector_size(16)));   // matches builtin's V4i

constexpr int Bb   = 4;
constexpr int Tt   = 2048;
constexpr int Cc   = 2048;
constexpr int Hh   = 16;
constexpr int Dd   = 128;      // C / H
constexpr int Mtot = Bb * Tt;  // 8192 tokens
constexpr int Nqkv = 3 * Cc;   // 6144
constexpr float SCALE = 0.08838834764831845f;  // 1/sqrt(128)

// Async global->LDS copy (gfx1250 ASYNCcnt path), guarded so absence of the
// builtin falls back to the known-good synchronous staging.
#if __has_builtin(__builtin_amdgcn_global_load_async_to_lds_b128)
#define HAVE_ASYNC_LDS 1
#define ASYNC_LDS_B128(gsrc, ldst)                                                   \
  __builtin_amdgcn_global_load_async_to_lds_b128(                                    \
      (__attribute__((address_space(1))) v4i_vs*)(uintptr_t)(const void*)(gsrc),     \
      (__attribute__((address_space(3))) v4i_vs*)(uint32_t)(uintptr_t)(void*)(ldst), \
      0, 0)
__device__ __forceinline__ void wait_async0() {
#if __has_builtin(__builtin_amdgcn_s_wait_asynccnt)
  __builtin_amdgcn_s_wait_asynccnt(0);
#else
  asm volatile("s_wait_asynccnt 0x0" ::: "memory");
#endif
}
#endif

// K index inside a 16-bit A-fragment (16x32) for VGPR-pair slot p and lane half.
__device__ __forceinline__ int kmapA(int p, int half) {
  return 16 * half + 2 * p;
}

__device__ __forceinline__ v8f wmma_bf16(v16bf a, v16bf b, v8f c) {
  return __builtin_amdgcn_wmma_f32_16x16x32_bf16(
      /*neg_a=*/false, a, /*neg_b=*/false, b,
      /*c_mod=*/(short)0, c, /*reuse_a=*/false, /*reuse_b=*/false);
}

__device__ __forceinline__ v8f v8f_zero() {
  v8f z = {0.f, 0.f, 0.f, 0.f, 0.f, 0.f, 0.f, 0.f};
  return z;
}

// ---------------------------------------------------------------- convert ---
__global__ void cvt_f32_bf16(const float* __restrict__ in, bf16* __restrict__ out, size_t n) {
  size_t i = (size_t)blockIdx.x * blockDim.x + threadIdx.x;
  size_t stride = (size_t)gridDim.x * blockDim.x;
  for (; i < n; i += stride) out[i] = (bf16)in[i];
}

// ------------------------------------------------------------------- GEMM ---
// Cout[m, n] = sum_k A[m, k] * Bw[n, k]     (Bw row-major: N x K)
// Block tile 256(M) x 128(N), 8 waves as 4(M) x 2(N), each wave 64x64 =
// 4x4 WMMA accumulators (16 WMMAs per 32-deep K step, 8 fragment loads).
// Global -> register -> LDS software pipeline hides HBM latency.
template <typename OutT>
__global__ __launch_bounds__(256) void gemm_bf16(const bf16* __restrict__ A,
                                                 const bf16* __restrict__ Bw,
                                                 OutT* __restrict__ Cout,
                                                 int M, int N, int K) {
  __shared__ bf16 lA[256 * 32];   // 16 KB
  __shared__ bf16 lB[128 * 32];   //  8 KB

  const int tid  = threadIdx.x;
  const int lane = tid & 31;
  const int wave = tid >> 5;
  const int half = lane >> 4;
  const int l16  = lane & 15;

  const int m0 = blockIdx.y * 256;
  const int n0 = blockIdx.x * 128;
  const int wm = (wave >> 1) * 64;   // 0,64,128,192
  const int wn = (wave & 1) * 64;    // 0,64

  v8f acc[4][4];
#pragma unroll
  for (int i = 0; i < 4; ++i)
#pragma unroll
    for (int j = 0; j < 4; ++j) acc[i][j] = v8f_zero();

  // Staging assignment: 16-bf16 (32B) chunks.
  const int arow = tid >> 1;
  const int aseg = (tid & 1) * 16;
  const int brow = tid >> 1;
  const int bseg = (tid & 1) * 16;

  uint4 sa0[2], sa1[2], sb[2];

  // ---- prologue: load k0 = 0 tile and commit to LDS
  {
    const uint4* p;
    p = reinterpret_cast<const uint4*>(A + (size_t)(m0 + arow) * K + aseg);
    sa0[0] = p[0]; sa0[1] = p[1];
    p = reinterpret_cast<const uint4*>(A + (size_t)(m0 + arow + 128) * K + aseg);
    sa1[0] = p[0]; sa1[1] = p[1];
    p = reinterpret_cast<const uint4*>(Bw + (size_t)(n0 + brow) * K + bseg);
    sb[0] = p[0]; sb[1] = p[1];
  }
  *reinterpret_cast<uint4*>(&lA[arow * 32 + aseg])             = sa0[0];
  *reinterpret_cast<uint4*>(&lA[arow * 32 + aseg + 8])         = sa0[1];
  *reinterpret_cast<uint4*>(&lA[(arow + 128) * 32 + aseg])     = sa1[0];
  *reinterpret_cast<uint4*>(&lA[(arow + 128) * 32 + aseg + 8]) = sa1[1];
  *reinterpret_cast<uint4*>(&lB[brow * 32 + bseg])             = sb[0];
  *reinterpret_cast<uint4*>(&lB[brow * 32 + bseg + 8])         = sb[1];
  __syncthreads();

#pragma unroll 1
  for (int k0 = 0; k0 < K; k0 += 32) {
    const bool more = (k0 + 32) < K;

    // ---- issue global loads for tile k0+32 while we compute tile k0
    if (more) {
      const int kn = k0 + 32;
      const uint4* p;
      p = reinterpret_cast<const uint4*>(A + (size_t)(m0 + arow) * K + kn + aseg);
      sa0[0] = p[0]; sa0[1] = p[1];
      p = reinterpret_cast<const uint4*>(A + (size_t)(m0 + arow + 128) * K + kn + aseg);
      sa1[0] = p[0]; sa1[1] = p[1];
      p = reinterpret_cast<const uint4*>(Bw + (size_t)(n0 + brow) * K + kn + bseg);
      sb[0] = p[0]; sb[1] = p[1];
      if (k0 + 64 < K) {
        __builtin_prefetch(A + (size_t)(m0 + arow) * K + k0 + 64 + aseg, 0, 0);
        __builtin_prefetch(Bw + (size_t)(n0 + brow) * K + k0 + 64 + bseg, 0, 0);
      }
    }

    // ---- B fragments for this wave's four 16-col tiles
    v16bf bfr[4];
#pragma unroll
    for (int j = 0; j < 4; ++j) {
      unsigned* bp = reinterpret_cast<unsigned*>(&bfr[j]);
      const int nrow = wn + j * 16 + l16;
#pragma unroll
      for (int p = 0; p < 8; ++p)
        bp[p] = *reinterpret_cast<const unsigned*>(&lB[nrow * 32 + 16 * half + 2 * p]);
    }

    // ---- A fragments + 16 WMMAs
#pragma unroll
    for (int i = 0; i < 4; ++i) {
      v16bf afr;
      unsigned* ap = reinterpret_cast<unsigned*>(&afr);
      const int mrow = wm + i * 16 + l16;
#pragma unroll
      for (int p = 0; p < 8; ++p)
        ap[p] = *reinterpret_cast<const unsigned*>(&lA[mrow * 32 + kmapA(p, half)]);
#pragma unroll
      for (int j = 0; j < 4; ++j)
        acc[i][j] = wmma_bf16(afr, bfr[j], acc[i][j]);
    }
    __syncthreads();

    // ---- commit staged registers to LDS for the next iteration
    if (more) {
      *reinterpret_cast<uint4*>(&lA[arow * 32 + aseg])             = sa0[0];
      *reinterpret_cast<uint4*>(&lA[arow * 32 + aseg + 8])         = sa0[1];
      *reinterpret_cast<uint4*>(&lA[(arow + 128) * 32 + aseg])     = sa1[0];
      *reinterpret_cast<uint4*>(&lA[(arow + 128) * 32 + aseg + 8]) = sa1[1];
      *reinterpret_cast<uint4*>(&lB[brow * 32 + bseg])             = sb[0];
      *reinterpret_cast<uint4*>(&lB[brow * 32 + bseg + 8])         = sb[1];
    }
    __syncthreads();
  }

  // ---- epilogue: C layout row = r + 8*half, col = l16
#pragma unroll
  for (int i = 0; i < 4; ++i)
#pragma unroll
    for (int j = 0; j < 4; ++j)
#pragma unroll
      for (int r = 0; r < 8; ++r) {
        const int mm = m0 + wm + i * 16 + r + 8 * half;
        const int nn = n0 + wn + j * 16 + l16;
        Cout[(size_t)mm * N + nn] = (OutT)acc[i][j][r];
      }
}

// --------------------------------------------------- RMSNorm + RoPE + split --
// One wave per (token m, head h). Each lane owns 4 contiguous d values.
__global__ __launch_bounds__(256) void rmsrope(const bf16* __restrict__ qkv,
                                               const float* __restrict__ cosT,
                                               const float* __restrict__ sinT,
                                               const float* __restrict__ gq,
                                               const float* __restrict__ gk,
                                               bf16* __restrict__ qT,
                                               bf16* __restrict__ kT,
                                               bf16* __restrict__ vT) {
  const int gw   = blockIdx.x * 8 + (threadIdx.x >> 5);
  const int lane = threadIdx.x & 31;
  const int h = gw & (Hh - 1);
  const int m = gw >> 4;          // token index
  const int b = m >> 11;          // T = 2048
  const int t = m & (Tt - 1);
  const int d0 = lane * 4;

  const size_t base = (size_t)m * Nqkv + h * Dd + d0;
  float q[4], k[4], v[4];
#pragma unroll
  for (int i = 0; i < 4; ++i) {
    q[i] = (float)qkv[base + 0 * Cc + i];
    k[i] = (float)qkv[base + 1 * Cc + i];
    v[i] = (float)qkv[base + 2 * Cc + i];
  }

  float sq = 0.f, sk = 0.f;
#pragma unroll
  for (int i = 0; i < 4; ++i) { sq += q[i] * q[i]; sk += k[i] * k[i]; }
#pragma unroll
  for (int off = 16; off >= 1; off >>= 1) {
    sq += __shfl_xor(sq, off, 32);
    sk += __shfl_xor(sk, off, 32);
  }
  const float rq = rsqrtf(sq * (1.f / Dd) + 1e-6f);
  const float rk = rsqrtf(sk * (1.f / Dd) + 1e-6f);
#pragma unroll
  for (int i = 0; i < 4; ++i) {
    q[i] *= rq * gq[d0 + i];
    k[i] *= rk * gk[d0 + i];
  }

  // rotate_half partner lives at d ^ 64 == lane ^ 16
  const float sgn = (lane < 16) ? -1.f : 1.f;
  float qo[4], ko[4];
#pragma unroll
  for (int i = 0; i < 4; ++i) {
    const float qp = __shfl_xor(q[i], 16, 32);
    const float kp = __shfl_xor(k[i], 16, 32);
    const float c = cosT[(size_t)t * Dd + d0 + i];
    const float s = sinT[(size_t)t * Dd + d0 + i];
    qo[i] = q[i] * c + sgn * qp * s;
    ko[i] = k[i] * c + sgn * kp * s;
  }

  const size_t ob = (((size_t)b * Hh + h) * Tt + t) * Dd + d0;
#pragma unroll
  for (int i = 0; i < 4; ++i) {
    qT[ob + i] = (bf16)qo[i];
    kT[ob + i] = (bf16)ko[i];
    vT[ob + i] = (bf16)v[i];
  }
}

// -------------------------------------------------------------- attention ---
// Flash-style. Block = 8 waves; each wave owns a 16-row q tile of one (b,h).
// K tile staged via async global->LDS (ASYNCcnt) when available; V staged
// transposed through VGPRs for contiguous B-fragment pairs.
__global__ __launch_bounds__(256) void attention(const bf16* __restrict__ Q,
                                                 const bf16* __restrict__ Kt,
                                                 const bf16* __restrict__ Vt,
                                                 bf16* __restrict__ Y) {
  __shared__ bf16 sK[32 * 128];    // (key, d)
  __shared__ bf16 sVt[128 * 32];   // (d, key)
  __shared__ bf16 sP[8][16 * 32];  // per-wave P scratch (row, key)

  const int tid  = threadIdx.x;
  const int lane = tid & 31;
  const int wave = tid >> 5;
  const int half = lane >> 4;
  const int l16  = lane & 15;

  const int bh = blockIdx.x >> 4;                     // b*H + h
  const int q0 = (blockIdx.x & 15) * 128 + wave * 16; // this wave's q rows
  const int b  = bh >> 4;
  const int h  = bh & (Hh - 1);

  // Resident Q A-fragments: 16 rows x 128 d = 4 fragments.
  v16bf qf[4];
  {
    const bf16* qrow = Q + ((size_t)bh * Tt + q0 + l16) * Dd;
#pragma unroll
    for (int ks = 0; ks < 4; ++ks) {
      unsigned* ap = reinterpret_cast<unsigned*>(&qf[ks]);
#pragma unroll
      for (int p = 0; p < 8; ++p)
        ap[p] = *reinterpret_cast<const unsigned*>(qrow + ks * 32 + kmapA(p, half));
    }
  }

  v8f o[8];
  float mrow[8], lrow[8];
#pragma unroll
  for (int r = 0; r < 8; ++r) { o[r] = v8f_zero(); mrow[r] = -3.0e38f; lrow[r] = 0.f; }

  const int srow = tid >> 3;         // 0..31 (key row)
  const int sseg = (tid & 7) * 16;   // d segment

#pragma unroll 1
  for (int kt = 0; kt < Tt / 32; ++kt) {
    // ---- stage K (row-major, async when available) and V (transposed)
    {
      const size_t gb = ((size_t)bh * Tt + kt * 32 + srow) * Dd + sseg;
#ifdef HAVE_ASYNC_LDS
      ASYNC_LDS_B128(Kt + gb,     &sK[srow * 128 + sseg]);
      ASYNC_LDS_B128(Kt + gb + 8, &sK[srow * 128 + sseg + 8]);
#else
      const uint4* ksrc = reinterpret_cast<const uint4*>(Kt + gb);
      uint4 ka = ksrc[0], kb = ksrc[1];
      *reinterpret_cast<uint4*>(&sK[srow * 128 + sseg])     = ka;
      *reinterpret_cast<uint4*>(&sK[srow * 128 + sseg + 8]) = kb;
#endif
      const uint4* vsrc = reinterpret_cast<const uint4*>(Vt + gb);
      union { uint4 u[2]; bf16 e[16]; } vv;
      vv.u[0] = vsrc[0]; vv.u[1] = vsrc[1];
#pragma unroll
      for (int i = 0; i < 16; ++i)
        sVt[(sseg + i) * 32 + srow] = vv.e[i];
#ifdef HAVE_ASYNC_LDS
      wait_async0();
#endif
    }
    __syncthreads();

    // ---- scores: 16x32 = two 16x16 tiles, K-dim 128 in 4 WMMA steps
    v8f s0 = v8f_zero(), s1 = v8f_zero();
#pragma unroll
    for (int ks = 0; ks < 4; ++ks) {
      v16bf b0, b1;
      unsigned* b0p = reinterpret_cast<unsigned*>(&b0);
      unsigned* b1p = reinterpret_cast<unsigned*>(&b1);
#pragma unroll
      for (int p = 0; p < 8; ++p) {
        const int dk = ks * 32 + 16 * half + 2 * p;
        b0p[p] = *reinterpret_cast<const unsigned*>(&sK[(l16)      * 128 + dk]);
        b1p[p] = *reinterpret_cast<const unsigned*>(&sK[(16 + l16) * 128 + dk]);
      }
      s0 = wmma_bf16(qf[ks], b0, s0);
      s1 = wmma_bf16(qf[ks], b1, s1);
    }

    // ---- online softmax (row stats live per accumulator register r)
    float p0[8], p1[8], alpha[8];
#pragma unroll
    for (int r = 0; r < 8; ++r) {
      const float a = s0[r] * SCALE;
      const float c = s1[r] * SCALE;
      float mx = fmaxf(a, c);
#pragma unroll
      for (int off = 1; off < 16; off <<= 1) mx = fmaxf(mx, __shfl_xor(mx, off, 32));
      const float mnew = fmaxf(mrow[r], mx);
      alpha[r] = __expf(mrow[r] - mnew);
      mrow[r]  = mnew;
      p0[r] = __expf(a - mnew);
      p1[r] = __expf(c - mnew);
      float rs = p0[r] + p1[r];
#pragma unroll
      for (int off = 1; off < 16; off <<= 1) rs += __shfl_xor(rs, off, 32);
      lrow[r] = lrow[r] * alpha[r] + rs;
    }
#pragma unroll
    for (int nt = 0; nt < 8; ++nt)
#pragma unroll
      for (int r = 0; r < 8; ++r) o[nt][r] *= alpha[r];

    // ---- re-layout P (C layout -> A fragment) through per-wave LDS scratch
#pragma unroll
    for (int r = 0; r < 8; ++r) {
      sP[wave][(r + 8 * half) * 32 + l16]      = (bf16)p0[r];
      sP[wave][(r + 8 * half) * 32 + 16 + l16] = (bf16)p1[r];
    }
    v16bf pf;
    {
      unsigned* pp = reinterpret_cast<unsigned*>(&pf);
#pragma unroll
      for (int p = 0; p < 8; ++p)
        pp[p] = *reinterpret_cast<const unsigned*>(&sP[wave][l16 * 32 + kmapA(p, half)]);
    }

    // ---- o += P (16x32) x V (32x128) as 8 WMMAs
#pragma unroll
    for (int nt = 0; nt < 8; ++nt) {
      v16bf vf;
      unsigned* vp = reinterpret_cast<unsigned*>(&vf);
#pragma unroll
      for (int p = 0; p < 8; ++p)
        vp[p] = *reinterpret_cast<const unsigned*>(&sVt[(nt * 16 + l16) * 32 + 16 * half + 2 * p]);
      o[nt] = wmma_bf16(pf, vf, o[nt]);
    }
    __syncthreads();
  }

  // ---- epilogue: Y is (B, T, C) bf16 for the projection GEMM
#pragma unroll
  for (int r = 0; r < 8; ++r) {
    const float inv = 1.0f / lrow[r];
    const int t = q0 + r + 8 * half;
    const size_t base = ((size_t)b * Tt + t) * Cc + h * Dd;
#pragma unroll
    for (int nt = 0; nt < 8; ++nt)
      Y[base + nt * 16 + l16] = (bf16)(o[nt][r] * inv);
  }
}

// ------------------------------------------------------------------ driver --
extern "C" void kernel_launch(void* const* d_in, const int* in_sizes, int n_in,
                              void* d_out, int out_size, void* d_ws, size_t ws_size,
                              hipStream_t stream) {
  const float* x      = (const float*)d_in[0];
  const float* cosT   = (const float*)d_in[1];
  const float* sinT   = (const float*)d_in[2];
  const float* w_qkv  = (const float*)d_in[3];
  const float* w_proj = (const float*)d_in[4];
  const float* g_q    = (const float*)d_in[5];
  const float* g_k    = (const float*)d_in[6];
  float* out = (float*)d_out;

  char* ws = (char*)d_ws;
  bf16* xb     = (bf16*)ws; ws += (size_t)Mtot * Cc * sizeof(bf16);
  bf16* wqkvb  = (bf16*)ws; ws += (size_t)Nqkv * Cc * sizeof(bf16);
  bf16* wprojb = (bf16*)ws; ws += (size_t)Cc * Cc * sizeof(bf16);
  bf16* qkvb   = (bf16*)ws; ws += (size_t)Mtot * Nqkv * sizeof(bf16);
  bf16* qT     = (bf16*)ws; ws += (size_t)Mtot * Cc * sizeof(bf16);
  bf16* kT     = (bf16*)ws; ws += (size_t)Mtot * Cc * sizeof(bf16);
  bf16* vT     = (bf16*)ws; ws += (size_t)Mtot * Cc * sizeof(bf16);
  bf16* yb     = (bf16*)ws; ws += (size_t)Mtot * Cc * sizeof(bf16);

  // 1) fp32 -> bf16 conversions
  cvt_f32_bf16<<<8192, 256, 0, stream>>>(x, xb, (size_t)Mtot * Cc);
  cvt_f32_bf16<<<8192, 256, 0, stream>>>(w_qkv, wqkvb, (size_t)Nqkv * Cc);
  cvt_f32_bf16<<<4096, 256, 0, stream>>>(w_proj, wprojb, (size_t)Cc * Cc);

  // 2) QKV projection: (8192 x 2048) x (6144 x 2048)^T
  gemm_bf16<bf16><<<dim3(Nqkv / 128, Mtot / 256), 256, 0, stream>>>(
      xb, wqkvb, qkvb, Mtot, Nqkv, Cc);

  // 3) RMSNorm + RoPE + transpose to (B, H, T, D)
  rmsrope<<<(Mtot * Hh) / 8, 256, 0, stream>>>(qkvb, cosT, sinT, g_q, g_k, qT, kT, vT);

  // 4) Attention
  attention<<<Bb * Hh * (Tt / 128), 256, 0, stream>>>(qT, kT, vT, yb);

  // 5) Output projection: (8192 x 2048) x (2048 x 2048)^T -> fp32 out
  gemm_bf16<float><<<dim3(Cc / 128, Mtot / 256), 256, 0, stream>>>(
      yb, wprojb, out, Mtot, Cc, Cc);
}
